// Model_8340826489316
// MI455X (gfx1250) — matrix operand order repaired and checked by reading, not compiled, not run on hardware
//
#include <hip/hip_runtime.h>
#include <math.h>

// ---------------------------------------------------------------------------
// Fully fused xLSTM forward for MI455X (gfx1250, wave32).
// One wave32 workgroup per batch element; all activations live in LDS.
// mLSTM attention uses V_WMMA_F32_16X16X32_BF16 (K=32 == head dim).
// v3: V stored transposed in LDS so all WMMA B-tiles load as ds_load_b128
//     (v2's strided u16 gather emitted long v_mov_b16 packing chains);
//     cnb zeroing hoisted out of the head loop.
// ---------------------------------------------------------------------------

#define DEV __device__ __forceinline__

typedef __attribute__((ext_vector_type(16))) __bf16 v16bf;
typedef __attribute__((ext_vector_type(8)))  float  v8f;

constexpr int SEQ    = 48;
constexpr int DMODEL = 18;
constexpr int MI     = 64;   // mLSTM inner dim
constexpr int NBATCH = 4096;

// ---------------- params ---------------------------------------------------
struct MParams {
  const float *ln_w, *proj_up, *conv_w, *conv_b;
  const float *q_w, *k_w, *v_w;
  const float *ig_w, *ig_b, *fg_w, *fg_b;
  const float *outnorm_w, *skip, *proj_down;
};
struct SParams {
  const float *ln1_w, *conv_w, *conv_b;
  const float *ig_w, *fg_w, *zg_w, *og_w;
  const float *rec_w, *rec_b, *gn_w, *ln2_w, *ff_up, *ff_down;
};
struct AllParams {
  MParams mb[6];
  SParams sb;
  const float *post_norm_w, *dense_w, *dense_b;
};

// ---------------- LDS layout ------------------------------------------------
struct SmemM {
  float xn [SEQ][20];                       // LN output (D=18 padded)
  float xm [SEQ][MI];                       // proj_up low half; later attn out
  float xcb[SEQ][MI];                       // conv+silu output
  alignas(16) unsigned short zb16[SEQ][MI]; // z (bf16)
  alignas(16) unsigned short qb [SEQ][MI];  // q (bf16)
  alignas(16) unsigned short kb [SEQ][MI];  // k (bf16)
  alignas(16) unsigned short vbT[MI][SEQ];  // v TRANSPOSED (bf16): [dim][seq]
  alignas(16) unsigned short cnb[SEQ][MI];  // normalized scores (bf16, K-padded)
  float gI[2][SEQ];                         // input gate (per head)
  float gF[2][SEQ];                         // log-sigmoid forget gate
  float gC[2][SEQ];                         // cumsum of gF
};
struct SmemS {
  float xn [SEQ][20];   // ln1 out; later reused as y; later ln2 out
  float xc [SEQ][20];   // conv+silu out
  float gts[SEQ][4][20];// i,f,z,o pre-activations (18 used of 20)
  float hbuf[SEQ][MI];  // FFN hidden
  float hstate[24];     // recurrent h state (18 used)
};
struct Smem {
  float xres[SEQ][20];  // residual stream (D=18 padded)
  union { SmemM m; SmemS s; } u;
};

// ---------------- small helpers --------------------------------------------
DEV unsigned short f2bf(float f) {
  unsigned u = __builtin_bit_cast(unsigned, f);
  u = u + 0x7FFFu + ((u >> 16) & 1u);
  return (unsigned short)(u >> 16);
}
DEV float bf2f(unsigned short h) {
  unsigned u = ((unsigned)h) << 16;
  return __builtin_bit_cast(float, u);
}
DEV float sigmoidf_(float x) { return 1.f / (1.f + __expf(-x)); }
DEV float siluf_(float x)    { return x * sigmoidf_(x); }
DEV float logsigf_(float x) {
  return (x >= 0.f) ? -log1pf(__expf(-x)) : (x - log1pf(__expf(x)));
}

union BfTile {
  v16bf bf;
  unsigned short u[16];
  uint4 q4[2];
};

DEV v8f wmma_bf16(const BfTile& a, const BfTile& b, v8f c) {
  return __builtin_amdgcn_wmma_f32_16x16x32_bf16(
      false, a.bf, false, b.bf, (short)0, c, false, false);
}

// A-matrix 16x32 bf16 loader (contiguous k order, same as the B loaders).
// lanes 0-15: M=lane, K = 0..15; lanes 16-31: M=lane-16, K = 16..31
DEV BfTile loadA16x32(const unsigned short* base, int ld, int row0, int col0, int lane) {
  BfTile t;
  int m = lane & 15;
  int kh = (lane & 16) ? 16 : 0;
  const unsigned short* p = base + (row0 + m) * ld + col0 + kh;
  t.q4[0] = *(const uint4*)(p);       // K = kh .. kh+7
  t.q4[1] = *(const uint4*)(p + 8);   // K = kh+8 .. kh+15
  return t;
}

// B-matrix 32x16 bf16 where B[k][n] = src[nrow0+n][col0+k] (i.e. B = K^T),
// contiguous k order: lanes 0-15 hold K=0..15 for N=lane, lanes 16-31 K=16..31.
DEV BfTile loadB_fromT(const unsigned short* base, int ld, int nrow0, int col0, int lane) {
  BfTile t;
  int n = lane & 15;
  int kh = (lane & 16) ? 16 : 0;
  const unsigned short* p = base + (nrow0 + n) * ld + col0 + kh;
  t.q4[0] = *(const uint4*)(p);
  t.q4[1] = *(const uint4*)(p + 8);
  return t;
}

// B-matrix 32x16 bf16 from transposed V storage: B[k][n] = vbT[colbase+n][krow0+k].
// Lanes whose K-range starts beyond SEQ return zeros (K=48 padded to 64).
DEV BfTile loadB_fromVT(const unsigned short* vbT, int colbase, int krow0, int lane) {
  BfTile t;
  int n = lane & 15;
  int kh = (lane & 16) ? 16 : 0;
  int k0 = krow0 + kh;
  if (k0 < SEQ) {
    const unsigned short* p = vbT + (colbase + n) * SEQ + k0;
    t.q4[0] = *(const uint4*)(p);
    t.q4[1] = *(const uint4*)(p + 8);
  } else {
    t.q4[0] = make_uint4(0u, 0u, 0u, 0u);
    t.q4[1] = make_uint4(0u, 0u, 0u, 0u);
  }
  return t;
}

// One 16-row tile of the causal gated-score matrix: computes qk via WMMA for
// column tiles 0..MT, applies the log-D gating with row-max/row-sum done via
// 16-lane xor-shuffles, writes normalized bf16 scores to cnb.
// MT is compile-time so every accumulator is a named VGPR (no stack arrays).
template <int MT>
DEV void score_tile(SmemM& M, int h, int col0, int lane,
                    const BfTile& bk0, const BfTile& bk1, const BfTile& bk2) {
  const float qscale = 0.17677669529663687f; // 1/sqrt(32)
  BfTile aq = loadA16x32(&M.qb[0][0], MI, MT * 16, col0, lane);
  v8f z8 = {};
  v8f a0 = wmma_bf16(aq, bk0, z8);
  v8f a1 = z8, a2 = z8;
  if constexpr (MT >= 1) a1 = wmma_bf16(aq, bk1, z8);
  if constexpr (MT >= 2) a2 = wmma_bf16(aq, bk2, z8);

  int coll = lane & 15;
  int rhi  = (lane >> 4) << 3;

  // r-independent per-column gate values (hoisted out of the row loop)
  float gc0 = M.gC[h][coll],      gi0 = M.gI[h][coll];
  float gc1 = 0.f, gi1 = 0.f, gc2 = 0.f, gi2 = 0.f;
  if constexpr (MT >= 1) { gc1 = M.gC[h][16 + coll]; gi1 = M.gI[h][16 + coll]; }
  if constexpr (MT >= 2) { gc2 = M.gC[h][32 + coll]; gi2 = M.gI[h][32 + coll]; }

#pragma unroll
  for (int r = 0; r < 8; ++r) {
    int rloc = r + rhi;
    int row  = MT * 16 + rloc;
    float cumr = M.gC[h][row];

    // tiles strictly left of the diagonal are always causal-valid;
    // only the diagonal tile (nt == MT) needs the mask.
    float ld0 = -1e30f, ld1 = -1e30f, ld2 = -1e30f;
    if constexpr (MT == 0) {
      ld0 = (coll <= rloc) ? (cumr - gc0 + gi0) : -1e30f;
    } else {
      ld0 = cumr - gc0 + gi0;
    }
    if constexpr (MT == 1) {
      ld1 = (coll <= rloc) ? (cumr - gc1 + gi1) : -1e30f;
    } else if constexpr (MT > 1) {
      ld1 = cumr - gc1 + gi1;
    }
    if constexpr (MT == 2) {
      ld2 = (coll <= rloc) ? (cumr - gc2 + gi2) : -1e30f;
    }

    float lmax = fmaxf(ld0, fmaxf(ld1, ld2));
#pragma unroll
    for (int m = 1; m < 16; m <<= 1) lmax = fmaxf(lmax, __shfl_xor(lmax, m, 32));

    float cv0 = a0[r] * qscale * __expf(ld0 - lmax);
    float cv1 = 0.f, cv2 = 0.f;
    float rs = cv0;
    if constexpr (MT >= 1) { cv1 = a1[r] * qscale * __expf(ld1 - lmax); rs += cv1; }
    if constexpr (MT >= 2) { cv2 = a2[r] * qscale * __expf(ld2 - lmax); rs += cv2; }
#pragma unroll
    for (int m = 1; m < 16; m <<= 1) rs += __shfl_xor(rs, m, 32);

    float norm = fmaxf(fabsf(rs), __expf(-lmax)) + 1e-6f;
    float inv  = 1.f / norm;
    M.cnb[row][coll] = f2bf(cv0 * inv);
    if constexpr (MT >= 1) M.cnb[row][16 + coll] = f2bf(cv1 * inv);
    if constexpr (MT >= 2) M.cnb[row][32 + coll] = f2bf(cv2 * inv);
  }
}

// ---------------- mLSTM block ----------------------------------------------
DEV void mlstm_block(Smem& S_, const MParams& p, int lane) {
  auto& M = S_.u.m;

  // 1. LayerNorm over D=18
  for (int s = lane; s < SEQ; s += 32) {
    float sum = 0.f, sq = 0.f;
    for (int d = 0; d < DMODEL; ++d) { float v = S_.xres[s][d]; sum += v; sq += v * v; }
    float mu = sum / DMODEL;
    float var = sq / DMODEL - mu * mu;
    float r = rsqrtf(var + 1e-5f);
    for (int d = 0; d < DMODEL; ++d)
      M.xn[s][d] = (S_.xres[s][d] - mu) * r * p.ln_w[d];
  }
  __syncthreads();

  // 2. proj_up: 18 -> 128 (xm | z)
  for (int i = lane; i < SEQ * 128; i += 32) {
    int s = i >> 7, j = i & 127;
    float acc = 0.f;
    for (int d = 0; d < DMODEL; ++d) acc += M.xn[s][d] * p.proj_up[d * 128 + j];
    if (j < MI) M.xm[s][j] = acc;
    else        M.zb16[s][j - MI] = f2bf(acc);
  }
  __syncthreads();

  // 3. causal depthwise conv (K=4) + silu
  for (int i = lane; i < SEQ * MI; i += 32) {
    int s = i >> 6, c = i & 63;
    float acc = p.conv_b[c];
#pragma unroll
    for (int t = 0; t < 4; ++t) {
      int ss = s - 3 + t;
      if (ss >= 0) acc += M.xm[ss][c] * p.conv_w[t * MI + c];
    }
    M.xcb[s][c] = siluf_(acc);
  }
  __syncthreads();

  // 4. block-diagonal 4x4 q,k,v + bf16 pack (v stored transposed)
  for (int i = lane; i < SEQ * MI; i += 32) {
    int s = i >> 6, c = i & 63;
    int hb = c >> 2, o = c & 3;
    float qa = 0.f, ka = 0.f, va = 0.f;
#pragma unroll
    for (int t = 0; t < 4; ++t) {
      float xcv = M.xcb[s][hb * 4 + t];
      float xmv = M.xm [s][hb * 4 + t];
      qa += xcv * p.q_w[hb * 16 + o * 4 + t];
      ka += xcv * p.k_w[hb * 16 + o * 4 + t];
      va += xmv * p.v_w[hb * 16 + o * 4 + t];
    }
    M.qb[s][c]  = f2bf(qa);
    M.kb[s][c]  = f2bf(ka);
    M.vbT[c][s] = f2bf(va);
  }
  __syncthreads();

  // 5. ig / fg gates (192 -> 2), then per-head logsigmoid cumsum
  for (int i = lane; i < SEQ * 2; i += 32) {
    int s = i >> 1, nh = i & 1;
    float ai = p.ig_b[nh], af = p.fg_b[nh];
    for (int d = 0; d < 3 * MI; ++d) {
      float v = (d < MI) ? bf2f(M.qb[s][d])
              : (d < 2 * MI) ? bf2f(M.kb[s][d - MI])
              : bf2f(M.vbT[d - 2 * MI][s]);
      ai += v * p.ig_w[d * 2 + nh];
      af += v * p.fg_w[d * 2 + nh];
    }
    M.gI[nh][s] = ai;
    M.gF[nh][s] = logsigf_(af);
  }
  __syncthreads();
  for (int i = lane; i < 2 * SEQ; i += 32) {
    int nh = i / SEQ, s = i % SEQ;
    float c = 0.f;
    for (int t = 0; t <= s; ++t) c += M.gF[nh][t];
    M.gC[nh][s] = c;
  }
  // zero normalized-score buffer once per block (upper-tri / K-pad region
  // stays zero; the lower-tri region is fully overwritten per head)
  {
    unsigned int* z = (unsigned int*)&M.cnb[0][0];
    for (int i = lane; i < SEQ * MI / 2; i += 32) z[i] = 0u;
  }
  __syncthreads();

  // 6. gated attention per head, via WMMA bf16 (K=32)
  for (int h = 0; h < 2; ++h) {
    int col0 = 32 * h;

    // K-tiles of the score GEMM, loaded once per head
    BfTile bk0 = loadB_fromT(&M.kb[0][0], MI, 0,  col0, lane);
    BfTile bk1 = loadB_fromT(&M.kb[0][0], MI, 16, col0, lane);
    BfTile bk2 = loadB_fromT(&M.kb[0][0], MI, 32, col0, lane);
    score_tile<0>(M, h, col0, lane, bk0, bk1, bk2);
    score_tile<1>(M, h, col0, lane, bk0, bk1, bk2);
    score_tile<2>(M, h, col0, lane, bk0, bk1, bk2);
    __syncthreads();

    // h = Cn @ V : 3x2 output tiles, K=48 as two K=32 WMMA (zero padded).
    // V B-tiles (from transposed storage -> ds_load_b128) hoisted out of mt.
    BfTile vb00 = loadB_fromVT(&M.vbT[0][0], col0,      0,  lane);
    BfTile vb01 = loadB_fromVT(&M.vbT[0][0], col0,      32, lane);
    BfTile vb10 = loadB_fromVT(&M.vbT[0][0], col0 + 16, 0,  lane);
    BfTile vb11 = loadB_fromVT(&M.vbT[0][0], col0 + 16, 32, lane);

    int coll = lane & 15;
    int rhi  = (lane >> 4) << 3;
#pragma unroll
    for (int mt = 0; mt < 3; ++mt) {
      BfTile a0 = loadA16x32(&M.cnb[0][0], MI, mt * 16, 0,  lane);
      BfTile a1 = loadA16x32(&M.cnb[0][0], MI, mt * 16, 32, lane);
      v8f hz = {};
      v8f h0 = wmma_bf16(a0, vb00, hz);
      h0 = wmma_bf16(a1, vb01, h0);
      v8f h1 = wmma_bf16(a0, vb10, hz);
      h1 = wmma_bf16(a1, vb11, h1);
#pragma unroll
      for (int r = 0; r < 8; ++r) {
        int row = mt * 16 + r + rhi;
        M.xm[row][col0 + coll]      = h0[r]; // reuse xm as attn output
        M.xm[row][col0 + 16 + coll] = h1[r];
      }
    }
    __syncthreads();
  }

  // 7. per-head LN(DH=32) + skip + silu(z) gate
  for (int i = lane; i < SEQ * 2; i += 32) {
    int s = i >> 1, h = i & 1;
    float sum = 0.f, sq = 0.f;
    for (int d = 0; d < 32; ++d) { float v = M.xm[s][h * 32 + d]; sum += v; sq += v * v; }
    float mu = sum * (1.f / 32.f);
    float var = sq * (1.f / 32.f) - mu * mu;
    float r = rsqrtf(var + 1e-5f);
    for (int d = 0; d < 32; ++d) {
      int c = h * 32 + d;
      float hn = (M.xm[s][c] - mu) * r * p.outnorm_w[c];
      M.xm[s][c] = (hn + p.skip[c] * M.xcb[s][c]) * siluf_(bf2f(M.zb16[s][c]));
    }
  }
  __syncthreads();

  // 8. proj_down 64 -> 18 + residual
  for (int i = lane; i < SEQ * DMODEL; i += 32) {
    int s = i / DMODEL, d = i % DMODEL;
    float acc = 0.f;
    for (int c = 0; c < MI; ++c) acc += M.xm[s][c] * p.proj_down[c * DMODEL + d];
    S_.xres[s][d] += acc;
  }
  __syncthreads();
}

// ---------------- sLSTM block ----------------------------------------------
DEV void slstm_block(Smem& S_, const SParams& p, int lane) {
  auto& Ss = S_.u.s;

  // ln1
  for (int s = lane; s < SEQ; s += 32) {
    float sum = 0.f, sq = 0.f;
    for (int d = 0; d < DMODEL; ++d) { float v = S_.xres[s][d]; sum += v; sq += v * v; }
    float mu = sum / DMODEL, var = sq / DMODEL - mu * mu;
    float r = rsqrtf(var + 1e-5f);
    for (int d = 0; d < DMODEL; ++d)
      Ss.xn[s][d] = (S_.xres[s][d] - mu) * r * p.ln1_w[d];
  }
  __syncthreads();

  // causal conv (K=4, D=18) + silu
  for (int i = lane; i < SEQ * DMODEL; i += 32) {
    int s = i / DMODEL, d = i % DMODEL;
    float acc = p.conv_b[d];
#pragma unroll
    for (int t = 0; t < 4; ++t) {
      int ss = s - 3 + t;
      if (ss >= 0) acc += Ss.xn[ss][d] * p.conv_w[t * DMODEL + d];
    }
    Ss.xc[s][d] = siluf_(acc);
  }
  __syncthreads();

  // headwise gates: i,f from xc; z,o from xn   (2 heads x 9x9)
  for (int i = lane; i < SEQ * 4 * DMODEL; i += 32) {
    int s = i / (4 * DMODEL);
    int rem = i % (4 * DMODEL);
    int g = rem / DMODEL, hd = rem % DMODEL;
    int h = hd / 9, ii = hd % 9;
    const float* W = (g == 0) ? p.ig_w : (g == 1) ? p.fg_w : (g == 2) ? p.zg_w : p.og_w;
    float acc = 0.f;
    for (int j = 0; j < 9; ++j) {
      float src = (g < 2) ? Ss.xc[s][h * 9 + j] : Ss.xn[s][h * 9 + j];
      acc += src * W[(h * 9 + ii) * 9 + j];
    }
    Ss.gts[s][g][hd] = acc;
  }
  if (lane < 24) Ss.hstate[lane] = 0.f;
  __syncthreads();

  // recurrence: 48 sequential steps, lanes 0..17 active (lane = head*9 + i)
  float cst = 0.f, nst = 0.f, mst = 0.f;
  int hh = lane / 9, ii = lane % 9;
  for (int t = 0; t < SEQ; ++t) {
    float hnew = 0.f;
    if (lane < 2 * 9) {
      float hj[9];
#pragma unroll
      for (int j = 0; j < 9; ++j) hj[j] = Ss.hstate[hh * 9 + j];
      float raw[4];
#pragma unroll
      for (int g = 0; g < 4; ++g) {
        float a = Ss.gts[t][g][lane] + p.rec_b[(hh * 4 + g) * 9 + ii];
        const float* R = &p.rec_w[((hh * 4 + g) * 9 + ii) * 9];
#pragma unroll
        for (int j = 0; j < 9; ++j) a += R[j] * hj[j];
        raw[g] = a;
      }
      float ir = raw[0], fr = raw[1], zr = raw[2], orr = raw[3];
      float lfm = mst + logsigf_(fr);
      float mnew = (t == 0) ? ir : fmaxf(ir, lfm);
      float igt = __expf(ir - mnew);
      float fgt = __expf(lfm - mnew);
      float cnew = fgt * cst + igt * tanhf(zr);
      float nnew = fgt * nst + igt;
      hnew = sigmoidf_(orr) * cnew / nnew;
      cst = cnew; nst = nnew; mst = mnew;
      Ss.xn[t][lane] = hnew;           // xn reused as y
    }
    __syncthreads();
    if (lane < 2 * 9) Ss.hstate[lane] = hnew;
    __syncthreads();
  }

  // group-norm over DH=9 per (s, head) + residual
  for (int i = lane; i < SEQ * 2; i += 32) {
    int s = i >> 1, h = i & 1;
    float sum = 0.f, sq = 0.f;
    for (int d = 0; d < 9; ++d) { float v = Ss.xn[s][h * 9 + d]; sum += v; sq += v * v; }
    float mu = sum / 9.f, var = sq / 9.f - mu * mu;
    float r = rsqrtf(var + 1e-5f);
    for (int d = 0; d < 9; ++d) {
      int c = h * 9 + d;
      S_.xres[s][c] += (Ss.xn[s][c] - mu) * r * p.gn_w[c];
    }
  }
  __syncthreads();

  // ln2 (overwrite xn)
  for (int s = lane; s < SEQ; s += 32) {
    float sum = 0.f, sq = 0.f;
    for (int d = 0; d < DMODEL; ++d) { float v = S_.xres[s][d]; sum += v; sq += v * v; }
    float mu = sum / DMODEL, var = sq / DMODEL - mu * mu;
    float r = rsqrtf(var + 1e-5f);
    for (int d = 0; d < DMODEL; ++d)
      Ss.xn[s][d] = (S_.xres[s][d] - mu) * r * p.ln2_w[d];
  }
  __syncthreads();

  // FFN: gelu(gate) * up, then 64 -> 18
  for (int i = lane; i < SEQ * MI; i += 32) {
    int s = i >> 6, j = i & 63;
    float g = 0.f, u = 0.f;
    for (int d = 0; d < DMODEL; ++d) {
      float v = Ss.xn[s][d];
      g += v * p.ff_up[d * 128 + j];
      u += v * p.ff_up[d * 128 + 64 + j];
    }
    float gl = 0.5f * g * (1.f + erff(g * 0.70710678118654752f));
    Ss.hbuf[s][j] = gl * u;
  }
  __syncthreads();
  for (int i = lane; i < SEQ * DMODEL; i += 32) {
    int s = i / DMODEL, d = i % DMODEL;
    float acc = 0.f;
    for (int j = 0; j < MI; ++j) acc += Ss.hbuf[s][j] * p.ff_down[j * DMODEL + d];
    S_.xres[s][d] += acc;
  }
  __syncthreads();
}

// ---------------- fused kernel ---------------------------------------------
__global__ __launch_bounds__(32) void xlstm_fused(const float* __restrict__ x,
                                                  AllParams P,
                                                  float* __restrict__ out) {
  __shared__ Smem sm;
  int lane = threadIdx.x;
  int b = blockIdx.x;

  for (int i = lane; i < SEQ * DMODEL; i += 32) {
    int s = i / DMODEL, d = i % DMODEL;
    sm.xres[s][d] = x[(size_t)b * SEQ * DMODEL + i];
  }
  __syncthreads();

  for (int bi = 0; bi < 7; ++bi) {
    if (bi == 1) slstm_block(sm, P.sb, lane);
    else         mlstm_block(sm, P.mb[(bi < 1) ? bi : bi - 1], lane);
  }

  // final LN + dense(18->1) + zero first 24 positions
  for (int s = lane; s < SEQ; s += 32) {
    float sum = 0.f, sq = 0.f;
    for (int d = 0; d < DMODEL; ++d) { float v = sm.xres[s][d]; sum += v; sq += v * v; }
    float mu = sum / DMODEL, var = sq / DMODEL - mu * mu;
    float r = rsqrtf(var + 1e-5f);
    float acc = P.dense_b[0];
    for (int d = 0; d < DMODEL; ++d)
      acc += (sm.xres[s][d] - mu) * r * P.post_norm_w[d] * P.dense_w[d];
    out[(size_t)b * SEQ + s] = (s < 24) ? 0.f : acc;
  }
}

// ---------------- host glue -------------------------------------------------
extern "C" void kernel_launch(void* const* d_in, const int* in_sizes, int n_in,
                              void* d_out, int out_size, void* d_ws, size_t ws_size,
                              hipStream_t stream) {
  (void)in_sizes; (void)n_in; (void)d_ws; (void)ws_size; (void)out_size;

  const float* x = (const float*)d_in[0];
  AllParams P;
  int idx = 1;
  int mi = 0;
  for (int bi = 0; bi < 7; ++bi) {
    if (bi == 1) {
      SParams& s = P.sb;
      s.ln1_w   = (const float*)d_in[idx++];
      s.conv_w  = (const float*)d_in[idx++];
      s.conv_b  = (const float*)d_in[idx++];
      s.ig_w    = (const float*)d_in[idx++];
      s.fg_w    = (const float*)d_in[idx++];
      s.zg_w    = (const float*)d_in[idx++];
      s.og_w    = (const float*)d_in[idx++];
      s.rec_w   = (const float*)d_in[idx++];
      s.rec_b   = (const float*)d_in[idx++];
      s.gn_w    = (const float*)d_in[idx++];
      s.ln2_w   = (const float*)d_in[idx++];
      s.ff_up   = (const float*)d_in[idx++];
      s.ff_down = (const float*)d_in[idx++];
    } else {
      MParams& m = P.mb[mi++];
      m.ln_w      = (const float*)d_in[idx++];
      m.proj_up   = (const float*)d_in[idx++];
      m.conv_w    = (const float*)d_in[idx++];
      m.conv_b    = (const float*)d_in[idx++];
      m.q_w       = (const float*)d_in[idx++];
      m.k_w       = (const float*)d_in[idx++];
      m.v_w       = (const float*)d_in[idx++];
      m.ig_w      = (const float*)d_in[idx++];
      m.ig_b      = (const float*)d_in[idx++];
      m.fg_w      = (const float*)d_in[idx++];
      m.fg_b      = (const float*)d_in[idx++];
      m.outnorm_w = (const float*)d_in[idx++];
      m.skip      = (const float*)d_in[idx++];
      m.proj_down = (const float*)d_in[idx++];
    }
  }
  P.post_norm_w = (const float*)d_in[idx++];
  P.dense_w     = (const float*)d_in[idx++];
  P.dense_b     = (const float*)d_in[idx++];

  xlstm_fused<<<dim3(NBATCH), dim3(32), 0, stream>>>(x, P, (float*)d_out);
}
